// TrajectoryLSTM_51616916963624
// MI455X (gfx1250) — matrix operand order repaired and checked by reading, not compiled, not run on hardware
//
#include <hip/hip_runtime.h>

// ---------------------------------------------------------------------------
// TrajectoryLSTM on MI455X (gfx1250): bf16 WMMA GEMMs, f32 accum/state.
// B=16384, T=8, H=256, PRED=12.
// One workgroup = 32 batch rows, 512 threads = 16 waves.
// Wave w owns hidden columns [w*16, w*16+16) of all 4 gates (4 N-tiles),
// times 2 M-tiles -> 8 accumulators; each B fragment is loaded once from L2
// and reused by both M-tiles (halves L2 traffic per FLOP vs M=16).
// LSTM pointwise update runs on in-register WMMA accumulators; c never
// leaves registers, h round-trips LDS as bf16 (A operand of next GEMM).
// ---------------------------------------------------------------------------

typedef __attribute__((ext_vector_type(16))) __bf16 v16bf;
typedef __attribute__((ext_vector_type(8)))  float  v8f;
typedef unsigned short u16;
typedef unsigned int   u32;

#define NB    16384
#define TSTEP 8
#define NH    256
#define NPRED 12
#define MROWS 32

__device__ __forceinline__ u16 f2bf(float f) {
    u32 u = __float_as_uint(f);
    u += 0x7FFFu + ((u >> 16) & 1u);   // round-to-nearest-even
    return (u16)(u >> 16);
}
__device__ __forceinline__ float sigf(float x) {
    return 1.0f / (1.0f + __expf(-x));
}
__device__ __forceinline__ float tanhf_(float x) {
    x = fminf(fmaxf(x, -15.0f), 15.0f);
    float e = __expf(2.0f * x);
    return (e - 1.0f) / (e + 1.0f);
}

// ---------------------------------------------------------------------------
// Pack f32 weight W (N rows x 256 cols, row-major; B[k][n] = W[n][k]) into
// bf16 WMMA B-fragment layout: [nT][kT][lane][16elem], 32 bytes per lane.
// 16-bit B 32x16 tile: K = (lane>>4)*8 + (e&7) + (e>>3)*16,
// column n = nT*16 + (lane & 15).
// ---------------------------------------------------------------------------
__global__ void pack_b_kernel(const float* __restrict__ W, u16* __restrict__ out, int N) {
    int idx = blockIdx.x * blockDim.x + threadIdx.x;
    int total = N * 256;
    if (idx >= total) return;
    int e    = idx & 15;
    int lane = (idx >> 4) & 31;
    int t    = idx >> 9;          // nT*8 + kT  (K=256 -> 8 kTiles)
    int kT   = t & 7;
    int nT   = t >> 3;
    int n = nT * 16 + (lane & 15);
    int k = kT * 32 + (lane >> 4) * 8 + (e & 7) + (e >> 3) * 16;
    out[idx] = f2bf(W[n * 256 + k]);
}

struct Params {
    const float* x;        // (B, T, 2)
    const float* wih0;     // l1_Wih0 (1024, 2) raw f32 (scalar path)
    const u16*   pIh1; const u16* pIh2; const u16* pIh3;   // packed Wih, layers 1..3
    const u16*   pHh0; const u16* pHh1; const u16* pHh2; const u16* pHh3;
    const float* bih0; const float* bhh0;
    const float* bih1; const float* bhh1;
    const float* bih2; const float* bhh2;
    const float* bih3; const float* bhh3;
    const u16*   pFc1;     // packed fc1_w (256x256)
    const float* fc1_b;    // (256)
    const float* fc2_w;    // (2,256) raw f32
    const float* fc2_b;    // (2)
    float*       out;      // (B, PRED, 2)
};

__global__ void __launch_bounds__(512)
traj_lstm_kernel(Params P) {
    extern __shared__ unsigned char smem[];
    u16*   hbase  = (u16*)smem;                               // 4 * 32 * 256 bf16 (64 KB)
    float* posbuf = (float*)(smem + 4 * MROWS * NH * 2);      // 32 x 2
    float* fcsum  = posbuf + MROWS * 2;                       // 32 x 2

    const int tid  = threadIdx.x;
    const int lane = tid & 31;
    const int w    = tid >> 5;       // wave id 0..15
    const int half = lane >> 4;      // 0 or 1
    const int nl   = lane & 15;
    const int col  = w * 16 + nl;    // hidden column owned by this lane
    const int b0   = blockIdx.x * MROWS;

    u16* hb[4] = { hbase, hbase + MROWS * NH, hbase + 2 * MROWS * NH, hbase + 3 * MROWS * NH };

    const v8f vzero = {};
    v8f c_reg[4][2];                 // [layer][Mtile]
#pragma unroll
    for (int l = 0; l < 4; ++l)
#pragma unroll
        for (int mt = 0; mt < 2; ++mt) c_reg[l][mt] = vzero;

    for (int i = tid; i < 4 * MROWS * NH; i += 512) hbase[i] = 0;
    if (tid < MROWS * 2) fcsum[tid] = 0.0f;
    __syncthreads();

    // --- A-fragment (16x32 bf16) for M-tile mt from LDS: two ds_load_b128 ---
    auto load_a = [&](const u16* src, int kT, int mt) -> v16bf {
        union { v16bf v; uint4 q[2]; } u;
        const u16* p = src + (mt * 16 + nl) * NH + kT * 32 + half * 8;
        u.q[0] = *(const uint4*)(p);        // K = half*8 .. +7
        u.q[1] = *(const uint4*)(p + 16);   // K = 16 + half*8 .. +7
        return u.v;
    };

    // --- K=256 GEMM into 4(gates) x 2(Mtiles); B loaded once per gate/kT ---
    auto gemm8 = [&](v8f (&acc)[4][2], const u16* asrc, const u16* bmat) {
        for (int kT = 0; kT < 8; ++kT) {
            v16bf a0 = load_a(asrc, kT, 0);
            v16bf a1 = load_a(asrc, kT, 1);
#pragma unroll
            for (int g = 0; g < 4; ++g) {
                const int nT = g * 16 + w;
                v16bf b = *((const v16bf*)bmat + (nT * 8 + kT) * 32 + lane);
                acc[g][0] = __builtin_amdgcn_wmma_f32_16x16x32_bf16(
                    false, a0, false, b, (short)0, acc[g][0], false, false);
                acc[g][1] = __builtin_amdgcn_wmma_f32_16x16x32_bf16(
                    false, a1, false, b, (short)0, acc[g][1], false, false);
            }
        }
    };

    // --- K=2 input contribution for stack-1 layer-0 (scalar FMAs) ---
    auto add_x = [&](v8f (&acc)[4][2]) {
#pragma unroll
        for (int g = 0; g < 4; ++g) {
            const int n = g * 256 + col;
            const float w0 = P.wih0[n * 2 + 0];
            const float w1 = P.wih0[n * 2 + 1];
#pragma unroll
            for (int mt = 0; mt < 2; ++mt)
#pragma unroll
                for (int e = 0; e < 8; ++e) {
                    const int m = mt * 16 + half * 8 + e;
                    acc[g][mt][e] += posbuf[m * 2 + 0] * w0 + posbuf[m * 2 + 1] * w1;
                }
        }
    };

    // --- LSTM pointwise on in-register accumulators; h -> LDS bf16 ---
    auto pointwise = [&](v8f (&acc)[4][2], v8f (&cst)[2], u16* hdst,
                         const float* bihm, const float* bhhm) {
        const float bi = bihm[0 * 256 + col] + bhhm[0 * 256 + col];
        const float bf = bihm[1 * 256 + col] + bhhm[1 * 256 + col];
        const float bg = bihm[2 * 256 + col] + bhhm[2 * 256 + col];
        const float bo = bihm[3 * 256 + col] + bhhm[3 * 256 + col];
#pragma unroll
        for (int mt = 0; mt < 2; ++mt)
#pragma unroll
            for (int e = 0; e < 8; ++e) {
                const int m = mt * 16 + half * 8 + e;
                const float gi = sigf(acc[0][mt][e] + bi);
                const float gf = sigf(acc[1][mt][e] + bf);
                const float gg = tanhf_(acc[2][mt][e] + bg);
                const float go = sigf(acc[3][mt][e] + bo);
                const float c  = gf * cst[mt][e] + gi * gg;
                cst[mt][e] = c;
                hdst[m * NH + col] = f2bf(go * tanhf_(c));
            }
    };

    // --- one LSTM cell step ---
    auto cell = [&](const u16* inbuf, const u16* pIhm, const u16* pHhm, u16* hbm,
                    const float* bihm, const float* bhhm, v8f (&cst)[2], bool xpath) {
        v8f acc[4][2];
#pragma unroll
        for (int g = 0; g < 4; ++g)
#pragma unroll
            for (int mt = 0; mt < 2; ++mt) acc[g][mt] = vzero;
        if (xpath) add_x(acc);
        else       gemm8(acc, inbuf, pIhm);
        gemm8(acc, hbm, pHhm);
        __syncthreads();                     // all reads of hbm done before overwrite
        pointwise(acc, cst, hbm, bihm, bhhm);
        __syncthreads();                     // new h visible to next cell
    };

    // ===================== encoder: T timesteps =====================
    for (int t = 0; t < TSTEP; ++t) {
        if (tid < MROWS * 2) {
            const int m = tid >> 1, d = tid & 1;
            posbuf[m * 2 + d] = P.x[(size_t)(b0 + m) * (TSTEP * 2) + t * 2 + d];
        }
        __syncthreads();
        cell(nullptr, nullptr, P.pHh0, hb[0], P.bih0, P.bhh0, c_reg[0], true);
        cell(hb[0],   P.pIh1,  P.pHh1, hb[1], P.bih1, P.bhh1, c_reg[1], false);
        cell(hb[1],   P.pIh2,  P.pHh2, hb[2], P.bih2, P.bhh2, c_reg[2], false);
        cell(hb[2],   P.pIh3,  P.pHh3, hb[3], P.bih3, P.bhh3, c_reg[3], false);
    }
    // posbuf now holds x[:, T-1, :] == initial pos

    // ===================== decoder: PRED steps ======================
    const float wd0 = P.fc2_w[0 * 256 + col];
    const float wd1 = P.fc2_w[1 * 256 + col];
    for (int p = 0; p < NPRED; ++p) {
        cell(nullptr, nullptr, P.pHh0, hb[0], P.bih0, P.bhh0, c_reg[0], true);
        cell(hb[0],   P.pIh1,  P.pHh1, hb[1], P.bih1, P.bhh1, c_reg[1], false);
        cell(hb[1],   P.pIh2,  P.pHh2, hb[2], P.bih2, P.bhh2, c_reg[2], false);
        cell(hb[2],   P.pIh3,  P.pHh3, hb[3], P.bih3, P.bhh3, c_reg[3], false);

        // fc1: relu(h2_1 @ fc1_w^T + fc1_b); wave w -> N-tile w, 2 M-tiles
        v8f fa[2] = { vzero, vzero };
        for (int kT = 0; kT < 8; ++kT) {
            v16bf a0 = load_a(hb[3], kT, 0);
            v16bf a1 = load_a(hb[3], kT, 1);
            v16bf b = *((const v16bf*)P.pFc1 + (w * 8 + kT) * 32 + lane);
            fa[0] = __builtin_amdgcn_wmma_f32_16x16x32_bf16(
                false, a0, false, b, (short)0, fa[0], false, false);
            fa[1] = __builtin_amdgcn_wmma_f32_16x16x32_bf16(
                false, a1, false, b, (short)0, fa[1], false, false);
        }

        // fc2: in-register relu, partial products reduced via LDS float atomics
        {
            const float fb = P.fc1_b[col];
#pragma unroll
            for (int mt = 0; mt < 2; ++mt)
#pragma unroll
                for (int e = 0; e < 8; ++e) {
                    const int m = mt * 16 + half * 8 + e;
                    const float v = fmaxf(fa[mt][e] + fb, 0.0f);
                    atomicAdd(&fcsum[m * 2 + 0], v * wd0);   // ds_add_f32
                    atomicAdd(&fcsum[m * 2 + 1], v * wd1);
                }
        }
        __syncthreads();

        // finalize: pos += delta, store prediction, reset reduction buffer
        if (tid < MROWS * 2) {
            const int m = tid >> 1, d = tid & 1;
            const float np = posbuf[m * 2 + d] + fcsum[m * 2 + d] + P.fc2_b[d];
            posbuf[m * 2 + d] = np;
            fcsum[m * 2 + d]  = 0.0f;
            P.out[((size_t)(b0 + m) * NPRED + p) * 2 + d] = np;
        }
        __syncthreads();
    }
}

// ---------------------------------------------------------------------------
extern "C" void kernel_launch(void* const* d_in, const int* in_sizes, int n_in,
                              void* d_out, int out_size, void* d_ws, size_t ws_size,
                              hipStream_t stream) {
    (void)in_sizes; (void)n_in; (void)out_size; (void)ws_size;

    const float* x       = (const float*)d_in[0];
    const float* l1_Wih0 = (const float*)d_in[1];
    const float* l1_Whh0 = (const float*)d_in[2];
    const float* l1_bih0 = (const float*)d_in[3];
    const float* l1_bhh0 = (const float*)d_in[4];
    const float* l1_Wih1 = (const float*)d_in[5];
    const float* l1_Whh1 = (const float*)d_in[6];
    const float* l1_bih1 = (const float*)d_in[7];
    const float* l1_bhh1 = (const float*)d_in[8];
    const float* l2_Wih0 = (const float*)d_in[9];
    const float* l2_Whh0 = (const float*)d_in[10];
    const float* l2_bih0 = (const float*)d_in[11];
    const float* l2_bhh0 = (const float*)d_in[12];
    const float* l2_Wih1 = (const float*)d_in[13];
    const float* l2_Whh1 = (const float*)d_in[14];
    const float* l2_bih1 = (const float*)d_in[15];
    const float* l2_bhh1 = (const float*)d_in[16];
    const float* fc1_w   = (const float*)d_in[17];
    const float* fc1_b   = (const float*)d_in[18];
    const float* fc2_w   = (const float*)d_in[19];
    const float* fc2_b   = (const float*)d_in[20];

    unsigned char* ws = (unsigned char*)d_ws;
    size_t off = 0;
    auto carve = [&](size_t bytes) -> u16* {
        u16* p = (u16*)(ws + off);
        off += (bytes + 255) & ~(size_t)255;
        return p;
    };
    const size_t big = (size_t)1024 * 256 * sizeof(u16);  // 512 KB each
    u16* pHh0 = carve(big);
    u16* pHh1 = carve(big);
    u16* pHh2 = carve(big);
    u16* pHh3 = carve(big);
    u16* pIh1 = carve(big);
    u16* pIh2 = carve(big);
    u16* pIh3 = carve(big);
    u16* pFc1 = carve((size_t)256 * 256 * sizeof(u16));   // 128 KB

    // pack weights into WMMA B layout (bf16)
    pack_b_kernel<<<1024, 256, 0, stream>>>(l1_Whh0, pHh0, 1024);
    pack_b_kernel<<<1024, 256, 0, stream>>>(l1_Whh1, pHh1, 1024);
    pack_b_kernel<<<1024, 256, 0, stream>>>(l2_Whh0, pHh2, 1024);
    pack_b_kernel<<<1024, 256, 0, stream>>>(l2_Whh1, pHh3, 1024);
    pack_b_kernel<<<1024, 256, 0, stream>>>(l1_Wih1, pIh1, 1024);
    pack_b_kernel<<<1024, 256, 0, stream>>>(l2_Wih0, pIh2, 1024);
    pack_b_kernel<<<1024, 256, 0, stream>>>(l2_Wih1, pIh3, 1024);
    pack_b_kernel<<<256,  256, 0, stream>>>(fc1_w,   pFc1, 256);

    Params P;
    P.x = x;
    P.wih0 = l1_Wih0;
    P.pIh1 = pIh1; P.pIh2 = pIh2; P.pIh3 = pIh3;
    P.pHh0 = pHh0; P.pHh1 = pHh1; P.pHh2 = pHh2; P.pHh3 = pHh3;
    P.bih0 = l1_bih0; P.bhh0 = l1_bhh0;
    P.bih1 = l1_bih1; P.bhh1 = l1_bhh1;
    P.bih2 = l2_bih0; P.bhh2 = l2_bhh0;
    P.bih3 = l2_bih1; P.bhh3 = l2_bhh1;
    P.pFc1 = pFc1;
    P.fc1_b = fc1_b;
    P.fc2_w = fc2_w;
    P.fc2_b = fc2_b;
    P.out = (float*)d_out;

    const size_t smem_bytes = (size_t)4 * MROWS * NH * 2   // h buffers (64 KB)
                            + (size_t)MROWS * 2 * 4        // posbuf
                            + (size_t)MROWS * 2 * 4;       // fcsum
    traj_lstm_kernel<<<NB / MROWS, 512, smem_bytes, stream>>>(P);
}
